// AttentionFusionBlock_1984274890904
// MI455X (gfx1250) — compile-verified, matrix operand order checked by reading, not run on hardware
//
#include <hip/hip_runtime.h>
#include <hip/hip_bf16.h>
#include <stdint.h>

typedef __attribute__((ext_vector_type(16))) _Float16 v16h;
typedef __attribute__((ext_vector_type(8)))  float    v8f;
typedef __attribute__((ext_vector_type(4)))  int      v4i;

union FragAB { v16h v; _Float16 h[16]; uint32_t u[8]; };

#define NTOK   772
#define NT     196
#define NS     576
#define DDIM   256
#define NHEAD  4
#define HD     64
#define NLAYER 12
#define NBATCH 32

// Probe for the CDNA5 async global->LDS copy path (ASYNCcnt-tracked DMA).
#if defined(__has_builtin)
# if __has_builtin(__builtin_amdgcn_global_load_async_to_lds_b128) && \
     __has_builtin(__builtin_amdgcn_s_wait_asynccnt)
#  define USE_ASYNC_LDS 1
# endif
#endif
#ifndef USE_ASYNC_LDS
# define USE_ASYNC_LDS 0
#endif

// Builtin signature (from hipcc diagnostic): (v4i AS1*, v4i AS3*, imm int, imm int)
#define GV4(p) ((__attribute__((address_space(1))) v4i*)(p))
#define SV4(p) ((__attribute__((address_space(3))) v4i*)(p))

// ---------------------------------------------------------------------------
// Generic f16 WMMA GEMM:  C[M,N] = A[M,K] @ W[N,K]^T + bias[N]   (row-major)
// Block tile 64(M) x 128(N) x 32(K), 256 threads = 8 waves (4x2 wave grid),
// each wave computes a 16x64 strip as 4 x (16x16) v_wmma_f32_16x16x32_f16.
// Double-buffered LDS; tiles staged with GLOBAL_LOAD_ASYNC_TO_LDS_B128 when
// available (overlaps DMA with WMMA), else synchronous b128 staging.
// Epilogue modes: 0 = f16 store, 1 = f32 residual add, 2 = exact GELU -> f16,
//                 3 = f32 store, 4 = transposed SAT store into d_out.
// ---------------------------------------------------------------------------
#define BM 64
#define BN 128
#define BK 32
#define LDT 40   // LDS row stride in halfs (80B, 16B aligned, pads banks)

__global__ __launch_bounds__(256)
void gemm_f16_wmma(const _Float16* __restrict__ A,
                   const _Float16* __restrict__ W,
                   const float* __restrict__ bias,
                   _Float16* __restrict__ Ch,
                   float* __restrict__ Cf,
                   int M, int N, int K, int mode)
{
  __shared__ _Float16 sA[2][BM * LDT];
  __shared__ _Float16 sB[2][BN * LDT];

  const int tid  = threadIdx.x;
  const int lane = tid & 31;
  const int wid  = tid >> 5;
  const int wM   = wid >> 1;          // 0..3
  const int wN   = wid & 1;           // 0..1
  const int m0   = blockIdx.y * BM;
  const int n0   = blockIdx.x * BN;
  const int l15  = lane & 15;
  const int kph  = (lane < 16) ? 0 : 8;

  v8f acc[4];
  #pragma unroll
  for (int i = 0; i < 4; i++)
    #pragma unroll
    for (int r = 0; r < 8; r++) acc[i][r] = 0.0f;

  const int arow = tid >> 2;
  const int aq   = (tid & 3) * 8;
  const int c0   = tid * 2;
  const int brow0 = c0 >> 2,       bq0 = (c0 & 3) * 8;
  const int brow1 = (c0 + 1) >> 2, bq1 = ((c0 + 1) & 3) * 8;

  // Stage one K-tile into LDS buffer `buf`.
  auto issue = [&](int k0, int buf) {
#if USE_ASYNC_LDS
    __builtin_amdgcn_global_load_async_to_lds_b128(
        GV4(A + (size_t)(m0 + arow) * K + k0 + aq),
        SV4(&sA[buf][arow * LDT + aq]), 0, 0);
    __builtin_amdgcn_global_load_async_to_lds_b128(
        GV4(W + (size_t)(n0 + brow0) * K + k0 + bq0),
        SV4(&sB[buf][brow0 * LDT + bq0]), 0, 0);
    __builtin_amdgcn_global_load_async_to_lds_b128(
        GV4(W + (size_t)(n0 + brow1) * K + k0 + bq1),
        SV4(&sB[buf][brow1 * LDT + bq1]), 0, 0);
#else
    *(uint4*)&sA[buf][arow * LDT + aq] =
        *(const uint4*)(A + (size_t)(m0 + arow) * K + k0 + aq);
    *(uint4*)&sB[buf][brow0 * LDT + bq0] =
        *(const uint4*)(W + (size_t)(n0 + brow0) * K + k0 + bq0);
    *(uint4*)&sB[buf][brow1 * LDT + bq1] =
        *(const uint4*)(W + (size_t)(n0 + brow1) * K + k0 + bq1);
#endif
  };

  const int nk = K / BK;
  issue(0, 0);

  for (int kt = 0; kt < nk; ++kt) {
    const int buf = kt & 1;
    if (kt + 1 < nk) {
      issue((kt + 1) * BK, buf ^ 1);     // prefetch next tile into other buffer
#if USE_ASYNC_LDS
      __builtin_amdgcn_s_wait_asynccnt(3);   // own tile-kt loads complete
#endif
      __builtin_prefetch(A + (size_t)(m0 + arow) * K + (kt + 1) * BK + aq, 0, 1);
    } else {
#if USE_ASYNC_LDS
      __builtin_amdgcn_s_wait_asynccnt(0);
#endif
    }
    __syncthreads();

    // --- A fragment (16x32, CDNA5 in-lane K striping) ---
    FragAB af;
    {
      const _Float16* pa = &sA[buf][(wM * 16 + l15) * LDT];
      #pragma unroll
      for (int j = 0; j < 4; j++) af.u[j]     = *(const uint32_t*)&pa[kph + 2 * j];
      #pragma unroll
      for (int j = 0; j < 4; j++) af.u[4 + j] = *(const uint32_t*)&pa[16 + kph + 2 * j];
    }
    // --- all 4 B fragments up front so WMMAs can issue back-to-back ---
    FragAB bf[4];
    #pragma unroll
    for (int nf = 0; nf < 4; nf++) {
      const _Float16* pb = &sB[buf][(wN * 64 + nf * 16 + l15) * LDT];
      #pragma unroll
      for (int j = 0; j < 4; j++) bf[nf].u[j]     = *(const uint32_t*)&pb[kph + 2 * j];
      #pragma unroll
      for (int j = 0; j < 4; j++) bf[nf].u[4 + j] = *(const uint32_t*)&pb[16 + kph + 2 * j];
    }
    #pragma unroll
    for (int nf = 0; nf < 4; nf++)
      acc[nf] = __builtin_amdgcn_wmma_f32_16x16x32_f16(
          false, af.v, false, bf[nf].v, (short)0, acc[nf], false, false);
    __syncthreads();   // all waves done reading `buf` before it is refilled
  }

  // --- epilogue (C layout: VGPR r -> row r (+8 for upper lane half)) ---
  const int rb = m0 + wM * 16 + ((lane >> 4) & 1) * 8;
  const int cb = n0 + wN * 64 + l15;
  #pragma unroll
  for (int nf = 0; nf < 4; nf++) {
    const int col = cb + nf * 16;
    const float bv = bias[col];
    #pragma unroll
    for (int r = 0; r < 8; r++) {
      const int row = rb + r;
      float v = acc[nf][r] + bv;
      if (mode == 0) {
        Ch[(size_t)row * N + col] = (_Float16)v;
      } else if (mode == 1) {
        size_t idx = (size_t)row * N + col;
        Cf[idx] = Cf[idx] + v;
      } else if (mode == 2) {
        v = 0.5f * v * (1.0f + erff(v * 0.70710678118654752f));
        Ch[(size_t)row * N + col] = (_Float16)v;
      } else if (mode == 3) {
        Cf[(size_t)row * N + col] = v;
      } else { // mode 4: (b,p,c) -> (b,c,p) SAT output
        int b = row / NS, p = row - b * NS;
        Cf[(size_t)b * (256 * NS) + (size_t)col * NS + p] = v;
      }
    }
  }
}

// ---------------------------------------------------------------------------
// Flash-attention with WMMA. One wave per (query-tile 16, head, batch).
// qkv layout (f16): [b*772+tok][768] with q at +0, k at +256, v at +512,
// head h occupying a 64-wide slice. Keys processed in tiles of 32 with
// online softmax (xor-shuffle row reductions over 16-lane half-groups).
// P@V goes through a 16x32 LDS bounce to convert C-layout -> A-layout.
// ---------------------------------------------------------------------------
#define KT 32

__global__ __launch_bounds__(32)
void attn_wmma(const _Float16* __restrict__ qkv,
               _Float16* __restrict__ att,
               int q_start, int q_len, int k_len, float scale)
{
  __shared__ _Float16 sK[KT * 72];
  __shared__ _Float16 sV[KT * 72];
  __shared__ _Float16 sP[16 * LDT];

  const int lane   = threadIdx.x;
  const int qt     = blockIdx.x;
  const int head   = blockIdx.y;
  const int b      = blockIdx.z;
  const int laneHi = (lane >> 4) & 1;
  const int l15    = lane & 15;
  const int kph    = laneHi ? 8 : 0;

  // Q fragments (two K=32 chunks of head-dim 64), loaded straight from global
  FragAB qf[2];
  {
    const int qtok = q_start + qt * 16 + l15;   // padded rows stay in-buffer
    const _Float16* qp = qkv + (size_t)(b * NTOK + qtok) * 768 + head * HD;
    #pragma unroll
    for (int c = 0; c < 2; c++) {
      #pragma unroll
      for (int j = 0; j < 4; j++) qf[c].u[j]     = *(const uint32_t*)&qp[c * 32 + kph + 2 * j];
      #pragma unroll
      for (int j = 0; j < 4; j++) qf[c].u[4 + j] = *(const uint32_t*)&qp[c * 32 + 16 + kph + 2 * j];
    }
  }

  v8f acc[4];
  float rmax[8], rsum[8];
  #pragma unroll
  for (int i = 0; i < 4; i++)
    #pragma unroll
    for (int r = 0; r < 8; r++) acc[i][r] = 0.0f;
  #pragma unroll
  for (int r = 0; r < 8; r++) { rmax[r] = -3.0e38f; rsum[r] = 0.0f; }

  const int nkt = (k_len + KT - 1) / KT;
  for (int kt = 0; kt < nkt; ++kt) {
    // --- stage K,V rows (one token row per lane) ---
    {
      const int ktok = kt * KT + lane;
      uint4 kz = {0, 0, 0, 0}, vz = {0, 0, 0, 0};
      #pragma unroll
      for (int q8 = 0; q8 < 8; q8++) {
        if (ktok < NTOK) {
          const _Float16* kp = qkv + (size_t)(b * NTOK + ktok) * 768 + 256 + head * HD;
          const _Float16* vp = kp + 256;
          kz = *(const uint4*)&kp[q8 * 8];
          vz = *(const uint4*)&vp[q8 * 8];
        }
        *(uint4*)&sK[lane * 72 + q8 * 8] = kz;
        *(uint4*)&sV[lane * 72 + q8 * 8] = vz;
      }
    }
    __syncthreads();

    // --- scores S = (Q @ K^T) * scale, masked past k_len ---
    v8f sfr[2];
    #pragma unroll
    for (int cf = 0; cf < 2; cf++) {
      v8f s;
      #pragma unroll
      for (int r = 0; r < 8; r++) s[r] = 0.0f;
      #pragma unroll
      for (int kc = 0; kc < 2; kc++) {
        FragAB bfr;
        const _Float16* pk = &sK[(cf * 16 + l15) * 72 + kc * 32];
        #pragma unroll
        for (int j = 0; j < 4; j++) bfr.u[j]     = *(const uint32_t*)&pk[kph + 2 * j];
        #pragma unroll
        for (int j = 0; j < 4; j++) bfr.u[4 + j] = *(const uint32_t*)&pk[16 + kph + 2 * j];
        s = __builtin_amdgcn_wmma_f32_16x16x32_f16(
            false, qf[kc].v, false, bfr.v, (short)0, s, false, false);
      }
      const int col = kt * KT + cf * 16 + l15;
      const float msk = (col < k_len) ? 0.0f : -3.0e38f;
      #pragma unroll
      for (int r = 0; r < 8; r++) s[r] = s[r] * scale + msk;
      sfr[cf] = s;
    }

    // --- online softmax ---
    float mnew[8], corr[8];
    #pragma unroll
    for (int r = 0; r < 8; r++) {
      float m = fmaxf(sfr[0][r], sfr[1][r]);
      for (int d = 1; d < 16; d <<= 1) m = fmaxf(m, __shfl_xor(m, d, 32));
      mnew[r] = fmaxf(rmax[r], m);
      corr[r] = expf(rmax[r] - mnew[r]);
    }
    #pragma unroll
    for (int cf = 0; cf < 2; cf++)
      #pragma unroll
      for (int r = 0; r < 8; r++)
        sfr[cf][r] = expf(sfr[cf][r] - mnew[r]);
    #pragma unroll
    for (int r = 0; r < 8; r++) {
      float ps = sfr[0][r] + sfr[1][r];
      for (int d = 1; d < 16; d <<= 1) ps += __shfl_xor(ps, d, 32);
      rsum[r] = rsum[r] * corr[r] + ps;
      rmax[r] = mnew[r];
    }
    #pragma unroll
    for (int nf = 0; nf < 4; nf++)
      #pragma unroll
      for (int r = 0; r < 8; r++) acc[nf][r] *= corr[r];

    // --- P (C layout) -> LDS -> A layout fragment ---
    #pragma unroll
    for (int cf = 0; cf < 2; cf++)
      #pragma unroll
      for (int r = 0; r < 8; r++)
        sP[(r + 8 * laneHi) * LDT + cf * 16 + l15] = (_Float16)sfr[cf][r];
    __syncthreads();

    FragAB pf;
    {
      const _Float16* pp = &sP[l15 * LDT];
      #pragma unroll
      for (int j = 0; j < 4; j++) pf.u[j]     = *(const uint32_t*)&pp[kph + 2 * j];
      #pragma unroll
      for (int j = 0; j < 4; j++) pf.u[4 + j] = *(const uint32_t*)&pp[16 + kph + 2 * j];
    }
    #pragma unroll
    for (int nf = 0; nf < 4; nf++) {
      FragAB vf;
      #pragma unroll
      for (int j = 0; j < 8; j++) {
        const int kk = (j < 4) ? (kph + 2 * (j & 3)) : (16 + kph + 2 * (j & 3));
        vf.h[2 * j]     = sV[kk * 72 + nf * 16 + l15];
        vf.h[2 * j + 1] = sV[(kk + 1) * 72 + nf * 16 + l15];
      }
      acc[nf] = __builtin_amdgcn_wmma_f32_16x16x32_f16(
          false, pf.v, false, vf.v, (short)0, acc[nf], false, false);
    }
    __syncthreads();
  }

  // --- normalize and store (f16, packed back into [tok][256] head slice) ---
  #pragma unroll
  for (int nf = 0; nf < 4; nf++) {
    const int col = nf * 16 + l15;
    #pragma unroll
    for (int r = 0; r < 8; r++) {
      const int qrel = qt * 16 + r + 8 * laneHi;
      if (qrel < q_len) {
        const float v = acc[nf][r] / rsum[r];
        att[(size_t)(b * NTOK + q_start + qrel) * DDIM + head * HD + col] = (_Float16)v;
      }
    }
  }
}

// ---------------------------------------------------------------------------
// Small glue kernels
// ---------------------------------------------------------------------------
__global__ __launch_bounds__(256)
void ln_f16(const float* __restrict__ h, const float* __restrict__ g,
            const float* __restrict__ bb, _Float16* __restrict__ out)
{
  __shared__ float red[256];
  const int t = blockIdx.x, c = threadIdx.x;
  const float x = h[(size_t)t * DDIM + c];
  red[c] = x; __syncthreads();
  for (int s = 128; s > 0; s >>= 1) { if (c < s) red[c] += red[c + s]; __syncthreads(); }
  const float mean = red[0] * (1.0f / DDIM);
  __syncthreads();
  const float d = x - mean;
  red[c] = d * d; __syncthreads();
  for (int s = 128; s > 0; s >>= 1) { if (c < s) red[c] += red[c + s]; __syncthreads(); }
  const float var = red[0] * (1.0f / DDIM);
  out[(size_t)t * DDIM + c] = (_Float16)(d * rsqrtf(var + 1e-6f) * g[c] + bb[c]);
}

__global__ __launch_bounds__(256)
void f32_to_f16(const float* __restrict__ s, _Float16* __restrict__ d, int n)
{
  const int i = blockIdx.x * 256 + threadIdx.x;
  if (i < n) d[i] = (_Float16)s[i];
}

__global__ __launch_bounds__(256)
void embed_kernel(const float* __restrict__ z, const float* __restrict__ x,
                  const float* __restrict__ pu, const float* __restrict__ ps,
                  float* __restrict__ h)
{
  const int t = blockIdx.x;                 // 0 .. B*772-1
  const int b = t / NTOK, tok = t % NTOK, c = threadIdx.x;
  float v;
  if (tok < NT) v = z[((size_t)b * DDIM + c) * NT + tok] + pu[(size_t)tok * DDIM + c];
  else { const int p = tok - NT;
         v = x[((size_t)b * DDIM + c) * NS + p] + ps[(size_t)p * DDIM + c]; }
  h[(size_t)t * DDIM + c] = v;
}

__global__ __launch_bounds__(256)
void extract_sat(const float* __restrict__ h, _Float16* __restrict__ o)
{
  const int t = blockIdx.x;                 // 0 .. B*576-1
  const int b = t / NS, p = t % NS, c = threadIdx.x;
  o[(size_t)t * DDIM + c] = (_Float16)h[(size_t)(b * NTOK + NT + p) * DDIM + c];
}

__global__ __launch_bounds__(256)
void store_uav(const float* __restrict__ h, float* __restrict__ o)
{
  const int t = blockIdx.x;                 // 0 .. B*196-1
  const int b = t / NT, p = t % NT, c = threadIdx.x;
  o[(size_t)b * DDIM * NT + (size_t)c * NT + p] = h[(size_t)(b * NTOK + p) * DDIM + c];
}

// ---------------------------------------------------------------------------
// Host orchestration
// ---------------------------------------------------------------------------
extern "C" void kernel_launch(void* const* d_in, const int* in_sizes, int n_in,
                              void* d_out, int out_size, void* d_ws, size_t ws_size,
                              hipStream_t stream)
{
  (void)in_sizes; (void)n_in; (void)out_size; (void)ws_size;

  const float* z       = (const float*)d_in[0];
  const float* x       = (const float*)d_in[1];
  const float* pos_uav = (const float*)d_in[2];
  const float* pos_sat = (const float*)d_in[3];
  const float* ln1_g   = (const float*)d_in[4];
  const float* ln1_b   = (const float*)d_in[5];
  const float* qkv_w   = (const float*)d_in[6];
  const float* qkv_b   = (const float*)d_in[7];
  const float* proj_w  = (const float*)d_in[8];
  const float* proj_b  = (const float*)d_in[9];
  const float* ln2_g   = (const float*)d_in[10];
  const float* ln2_b   = (const float*)d_in[11];
  const float* fc1_w   = (const float*)d_in[12];
  const float* fc1_b   = (const float*)d_in[13];
  const float* fc2_w   = (const float*)d_in[14];
  const float* fc2_b   = (const float*)d_in[15];
  const float* out_w0  = (const float*)d_in[16];
  const float* out_b0  = (const float*)d_in[17];
  const float* out_w1  = (const float*)d_in[18];
  const float* out_b1  = (const float*)d_in[19];
  const float* out_w2  = (const float*)d_in[20];
  const float* out_b2  = (const float*)d_in[21];

  const int Mtok = NBATCH * NTOK;   // 24704 (=386*64)
  const int Msat = NBATCH * NS;     // 18432 (=288*64)

  char* ws = (char*)d_ws;
  size_t off = 0;
  auto alloc = [&](size_t bytes) -> char* {
    char* p = ws + off;
    off += (bytes + 255) & ~(size_t)255;
    return p;
  };

  float*    h     = (float*)    alloc((size_t)Mtok * DDIM * 4);
  _Float16* xn    = (_Float16*) alloc((size_t)Mtok * DDIM * 2);
  _Float16* qkv   = (_Float16*) alloc((size_t)Mtok * 768  * 2);
  _Float16* att   = (_Float16*) alloc((size_t)Mtok * DDIM * 2);
  _Float16* u1    = (_Float16*) alloc((size_t)Mtok * 1024 * 2);
  _Float16* wqkv  = (_Float16*) alloc((size_t)NLAYER * 768  * 256 * 2);
  _Float16* wproj = (_Float16*) alloc((size_t)NLAYER * 256  * 256 * 2);
  _Float16* wfc1  = (_Float16*) alloc((size_t)NLAYER * 1024 * 256 * 2);
  _Float16* wfc2  = (_Float16*) alloc((size_t)NLAYER * 256  * 1024 * 2);
  _Float16* w0h   = (_Float16*) alloc((size_t)768 * 256 * 2);
  _Float16* w1h   = (_Float16*) alloc((size_t)512 * 768 * 2);
  _Float16* w2h   = (_Float16*) alloc((size_t)256 * 512 * 2);
  // SAT-head activations reuse big buffers (layer loop is done by then)
  _Float16* s0   = u1;
  _Float16* s1   = u1 + (size_t)Msat * 768;
  _Float16* hsat = xn;

  auto conv = [&](const float* s, _Float16* d, size_t n) {
    f32_to_f16<<<dim3((unsigned)((n + 255) / 256)), dim3(256), 0, stream>>>(s, d, (int)n);
  };
  conv(qkv_w,  wqkv,  (size_t)NLAYER * 768  * 256);
  conv(proj_w, wproj, (size_t)NLAYER * 256  * 256);
  conv(fc1_w,  wfc1,  (size_t)NLAYER * 1024 * 256);
  conv(fc2_w,  wfc2,  (size_t)NLAYER * 256  * 1024);
  conv(out_w0, w0h,   (size_t)768 * 256);
  conv(out_w1, w1h,   (size_t)512 * 768);
  conv(out_w2, w2h,   (size_t)256 * 512);

  embed_kernel<<<dim3(Mtok), dim3(256), 0, stream>>>(z, x, pos_uav, pos_sat, h);

  const float scale = 0.125f;   // hd^-0.5 = 1/8
  for (int l = 0; l < NLAYER; l++) {
    ln_f16<<<dim3(Mtok), dim3(256), 0, stream>>>(h, ln1_g + l * 256, ln1_b + l * 256, xn);
    gemm_f16_wmma<<<dim3(768 / BN, Mtok / BM), 256, 0, stream>>>(
        xn, wqkv + (size_t)l * 768 * 256, qkv_b + l * 768, qkv, nullptr, Mtok, 768, 256, 0);
    attn_wmma<<<dim3(13, NHEAD, NBATCH), 32, 0, stream>>>(qkv, att, 0,  NT, NT,   scale);
    attn_wmma<<<dim3(36, NHEAD, NBATCH), 32, 0, stream>>>(qkv, att, NT, NS, NTOK, scale);
    gemm_f16_wmma<<<dim3(256 / BN, Mtok / BM), 256, 0, stream>>>(
        att, wproj + (size_t)l * 256 * 256, proj_b + l * 256, nullptr, h, Mtok, 256, 256, 1);
    ln_f16<<<dim3(Mtok), 256, 0, stream>>>(h, ln2_g + l * 256, ln2_b + l * 256, xn);
    gemm_f16_wmma<<<dim3(1024 / BN, Mtok / BM), 256, 0, stream>>>(
        xn, wfc1 + (size_t)l * 1024 * 256, fc1_b + l * 1024, u1, nullptr, Mtok, 1024, 256, 2);
    gemm_f16_wmma<<<dim3(256 / BN, Mtok / BM), 256, 0, stream>>>(
        u1, wfc2 + (size_t)l * 256 * 1024, fc2_b + l * 256, nullptr, h, Mtok, 256, 1024, 1);
  }

  // SAT head MLP: 256 -> 768 -> 512 -> 256, last GEMM stores transposed to d_out
  extract_sat<<<dim3(Msat), dim3(256), 0, stream>>>(h, hsat);
  gemm_f16_wmma<<<dim3(768 / BN, Msat / BM), 256, 0, stream>>>(
      hsat, w0h, out_b0, s0, nullptr, Msat, 768, 256, 0);
  gemm_f16_wmma<<<dim3(512 / BN, Msat / BM), 256, 0, stream>>>(
      s0, w1h, out_b1, s1, nullptr, Msat, 512, 768, 0);
  gemm_f16_wmma<<<dim3(256 / BN, Msat / BM), 256, 0, stream>>>(
      s1, w2h, out_b2, nullptr, (float*)d_out, Msat, 256, 512, 4);

  // UAV branch: transpose-store h[:, :196] as (B,256,14,14)
  store_uav<<<dim3(NBATCH * NT), 256, 0, stream>>>(
      h, (float*)d_out + (size_t)NBATCH * DDIM * NS);
}
